// MoistureLSTM_1529008357503
// MI455X (gfx1250) — matrix operand order repaired and checked by reading, not compiled, not run on hardware
//
#include <hip/hip_runtime.h>
#include <hip/hip_bf16.h>

typedef __attribute__((ext_vector_type(16))) _Float16 v16h;
typedef __attribute__((ext_vector_type(8)))  float    v8f;
typedef __attribute__((ext_vector_type(2)))  float    v2f;
typedef __attribute__((ext_vector_type(4)))  float    f32x4;

#define HDIM   64
#define FDIM   4
#define TSTEPS 512

__device__ __forceinline__ float fast_sig(float x) {
    float e = __expf(-x);
    return __builtin_amdgcn_rcpf(1.0f + e);
}
__device__ __forceinline__ float fast_tanh(float x) {
    float e = __expf(2.0f * x);                 // saturates correctly at +/-inf
    return 1.0f - 2.0f * __builtin_amdgcn_rcpf(e + 1.0f);
}

// B fragment (16x16x32 f16 WMMA): B is 32x16 (KxN). W stored row-major f16 [N][64].
// column N=ln; lanes 0-15: K=0..7 (v0-3), K=16..23 (v4-7); lanes 16-31: K=8..15, K=24..31.
__device__ __forceinline__ v16h load_bfrag(const _Float16* W, int n, int ks, int lh) {
    const _Float16* pp = W + n * HDIM + ks * 32 + lh * 8;
    v16h r;
    ((f32x4*)&r)[0] = *(const f32x4*)pp;
    ((f32x4*)&r)[1] = *(const f32x4*)(pp + 16);
    return r;
}

// A fragment (16x32 f16): row M=ln in lane; lanes0-15: K=0..7 (v0-3), K=16..23 (v4-7);
// lanes16-31: K=8..15, K=24..31. Hbuf row-major f16 [16][64].
__device__ __forceinline__ v16h load_afrag(const _Float16* Hbuf, int ks, int ln, int lh) {
    const _Float16* pp = Hbuf + ln * HDIM + ks * 32 + lh * 8;
    v16h r;
    ((f32x4*)&r)[0] = *(const f32x4*)pp;
    ((f32x4*)&r)[1] = *(const f32x4*)(pp + 16);
    return r;
}

__global__ void __launch_bounds__(64) lstm2_fused_wmma_kernel(
    const float* __restrict__ x,
    const float* __restrict__ Wih0, const float* __restrict__ Whh0,
    const float* __restrict__ bih0, const float* __restrict__ bhh0,
    const float* __restrict__ Wih1, const float* __restrict__ Whh1,
    const float* __restrict__ bih1, const float* __restrict__ bhh1,
    const float* __restrict__ Wh1,  const float* __restrict__ bh1,
    const float* __restrict__ Wh2,  const float* __restrict__ bh2,
    float* __restrict__ out)
{
    extern __shared__ char smem[];
    _Float16* sWhh0 = (_Float16*)smem;            // 256*64 f16 = 32KB
    _Float16* sWih1 = sWhh0 + 256 * HDIM;         // 32KB
    _Float16* sWhh1 = sWih1 + 256 * HDIM;         // 32KB
    _Float16* sH0   = sWhh1 + 256 * HDIM;         // 16*64 f16 = 2KB
    _Float16* sH1   = sH0 + 16 * HDIM;            // 2KB

    const int tid  = threadIdx.x;
    const int lane = tid & 31;
    const int p    = tid >> 5;        // wave parity: owns hidden units [32p, 32p+32)
    const int ln   = lane & 15;
    const int lh   = lane >> 4;
    const int rowBase = blockIdx.x * 16;

    // ---- prologue: stage f32 weights -> f16 LDS, zero h state ----
    for (int i = tid; i < 256 * HDIM; i += blockDim.x) {
        sWhh0[i] = (_Float16)Whh0[i];
        sWih1[i] = (_Float16)Wih1[i];
        sWhh1[i] = (_Float16)Whh1[i];
    }
    for (int i = tid; i < 16 * HDIM; i += blockDim.x) {
        sH0[i] = (_Float16)0.f;
        sH1[i] = (_Float16)0.f;
    }

    // per-wave constants: combined per-lane bias scalars and Wih0 B-fragments.
    // NOTE: biases are kept as SCALARS (1 VGPR each) and added in the elementwise
    // phase; splatting them into v8f C operands caused 128 VGPRs of spill.
    float bias0v[4][2], bias1v[4][2];
    v2f   bW0[4][2];   // B frag for 16x16x4 f32 wmma: lane holds Wih0[n][2lh], [n][2lh+1]
#pragma unroll
    for (int gi = 0; gi < 4; ++gi)
#pragma unroll
        for (int q = 0; q < 2; ++q) {
            int gt = gi * 4 + 2 * p + q;      // global 16-col gate tile (0..15)
            int n  = gt * 16 + ln;            // gate column
            bias0v[gi][q] = bih0[n] + bhh0[n];
            bias1v[gi][q] = bih1[n] + bhh1[n];
            bW0[gi][q] = *(const v2f*)(Wih0 + n * FDIM + 2 * lh);
        }

    v8f c0v[2] = {};                          // layer-0 cell state (16 rows x 32 cols)
    v8f c1v[2] = {};                          // layer-1 cell state
    v8f acc[4][2];

    __syncthreads();

    const float* xlane = x + ((size_t)(rowBase + ln) * TSTEPS) * FDIM + 2 * lh;

#pragma unroll 1
    for (int t = 0; t < TSTEPS; ++t) {
        // ================= layer 0 =================
        // A frag for 16x4 f32 wmma: VGPR0=K(2lh), VGPR1=K(2lh+1), row M=ln
        v2f xa = *(const v2f*)(xlane + t * FDIM);
        int tp = (t + 16 < TSTEPS) ? (t + 16) : t;
        __builtin_prefetch(xlane + tp * FDIM, 0, 1);

        v8f zc = {0.f, 0.f, 0.f, 0.f, 0.f, 0.f, 0.f, 0.f};

#pragma unroll
        for (int gi = 0; gi < 4; ++gi)
#pragma unroll
            for (int q = 0; q < 2; ++q) {
                acc[gi][q] = __builtin_amdgcn_wmma_f32_16x16x4_f32(
                    false, xa, false, bW0[gi][q], (short)0, zc, false, false);
            }

        v16h a0 = load_afrag(sH0, 0, ln, lh);   // h0_{t-1}, K=0..31
        v16h a1 = load_afrag(sH0, 1, ln, lh);   // K=32..63
#pragma unroll
        for (int gi = 0; gi < 4; ++gi)
#pragma unroll
            for (int q = 0; q < 2; ++q) {
                int n = (gi * 4 + 2 * p + q) * 16 + ln;
                v16h b0 = load_bfrag(sWhh0, n, 0, lh);
                acc[gi][q] = __builtin_amdgcn_wmma_f32_16x16x32_f16(
                    false, a0, false, b0, (short)0, acc[gi][q], false, false);
                v16h b1 = load_bfrag(sWhh0, n, 1, lh);
                acc[gi][q] = __builtin_amdgcn_wmma_f32_16x16x32_f16(
                    false, a1, false, b1, (short)0, acc[gi][q], false, false);
            }

        __syncthreads();   // S1: all waves done reading h0_{t-1}

#pragma unroll
        for (int q = 0; q < 2; ++q) {
            int col = (2 * p + q) * 16 + ln;
#pragma unroll
            for (int r = 0; r < 8; ++r) {
                float ii = fast_sig (acc[0][q][r] + bias0v[0][q]);
                float ff = fast_sig (acc[1][q][r] + bias0v[1][q]);
                float gg = fast_tanh(acc[2][q][r] + bias0v[2][q]);
                float oo = fast_sig (acc[3][q][r] + bias0v[3][q]);
                float c  = ff * c0v[q][r] + ii * gg;
                c0v[q][r] = c;
                float h  = oo * fast_tanh(c);
                sH0[(r + 8 * lh) * HDIM + col] = (_Float16)h;   // D-layout -> row-major
            }
        }

        __syncthreads();   // S2: h0_t visible to both waves

        // ================= layer 1 =================
        a0 = load_afrag(sH0, 0, ln, lh);        // h0_t (input term)
        a1 = load_afrag(sH0, 1, ln, lh);
#pragma unroll
        for (int gi = 0; gi < 4; ++gi)
#pragma unroll
            for (int q = 0; q < 2; ++q) {
                int n = (gi * 4 + 2 * p + q) * 16 + ln;
                v16h b0 = load_bfrag(sWih1, n, 0, lh);
                v8f cc = __builtin_amdgcn_wmma_f32_16x16x32_f16(
                    false, a0, false, b0, (short)0, zc, false, false);
                v16h b1 = load_bfrag(sWih1, n, 1, lh);
                acc[gi][q] = __builtin_amdgcn_wmma_f32_16x16x32_f16(
                    false, a1, false, b1, (short)0, cc, false, false);
            }

        v16h h1a0 = load_afrag(sH1, 0, ln, lh); // h1_{t-1} (recurrent term)
        v16h h1a1 = load_afrag(sH1, 1, ln, lh);
#pragma unroll
        for (int gi = 0; gi < 4; ++gi)
#pragma unroll
            for (int q = 0; q < 2; ++q) {
                int n = (gi * 4 + 2 * p + q) * 16 + ln;
                v16h b0 = load_bfrag(sWhh1, n, 0, lh);
                acc[gi][q] = __builtin_amdgcn_wmma_f32_16x16x32_f16(
                    false, h1a0, false, b0, (short)0, acc[gi][q], false, false);
                v16h b1 = load_bfrag(sWhh1, n, 1, lh);
                acc[gi][q] = __builtin_amdgcn_wmma_f32_16x16x32_f16(
                    false, h1a1, false, b1, (short)0, acc[gi][q], false, false);
            }

        __syncthreads();   // S3: all waves done reading h1_{t-1}

#pragma unroll
        for (int q = 0; q < 2; ++q) {
            int col = (2 * p + q) * 16 + ln;
#pragma unroll
            for (int r = 0; r < 8; ++r) {
                float ii = fast_sig (acc[0][q][r] + bias1v[0][q]);
                float ff = fast_sig (acc[1][q][r] + bias1v[1][q]);
                float gg = fast_tanh(acc[2][q][r] + bias1v[2][q]);
                float oo = fast_sig (acc[3][q][r] + bias1v[3][q]);
                float c  = ff * c1v[q][r] + ii * gg;
                c1v[q][r] = c;
                float h  = oo * fast_tanh(c);
                sH1[(r + 8 * lh) * HDIM + col] = (_Float16)h;
            }
        }
        // next read of sH1 is after S2 of the following iteration -> no barrier needed here
    }

    __syncthreads();

    // ---- head: Linear(64,32) -> ReLU -> Linear(32,1), one lane per batch row ----
    if (tid < 16) {
        int m = tid;
        float y = bh2[0];
        for (int u = 0; u < 32; ++u) {
            float s = bh1[u];
#pragma unroll 8
            for (int k = 0; k < HDIM; ++k)
                s += (float)sH1[m * HDIM + k] * Wh1[u * HDIM + k];
            y += fmaxf(s, 0.0f) * Wh2[u];
        }
        out[rowBase + m] = y;
    }
}

extern "C" void kernel_launch(void* const* d_in, const int* in_sizes, int n_in,
                              void* d_out, int out_size, void* d_ws, size_t ws_size,
                              hipStream_t stream) {
    const float* x    = (const float*)d_in[0];
    const float* Wih0 = (const float*)d_in[1];
    const float* Whh0 = (const float*)d_in[2];
    const float* bih0 = (const float*)d_in[3];
    const float* bhh0 = (const float*)d_in[4];
    const float* Wih1 = (const float*)d_in[5];
    const float* Whh1 = (const float*)d_in[6];
    const float* bih1 = (const float*)d_in[7];
    const float* bhh1 = (const float*)d_in[8];
    const float* Wh1  = (const float*)d_in[9];
    const float* bh1  = (const float*)d_in[10];
    const float* Wh2  = (const float*)d_in[11];
    const float* bh2  = (const float*)d_in[12];
    float* out = (float*)d_out;

    const int B = in_sizes[0] / (TSTEPS * FDIM);   // 2048
    const size_t smem = (size_t)(3 * 256 * HDIM + 2 * 16 * HDIM) * sizeof(_Float16); // 100KB

    hipFuncSetAttribute((const void*)lstm2_fused_wmma_kernel,
                        hipFuncAttributeMaxDynamicSharedMemorySize, (int)smem);

    lstm2_fused_wmma_kernel<<<dim3(B / 16), dim3(64), smem, stream>>>(
        x, Wih0, Whh0, bih0, bhh0, Wih1, Whh1, bih1, bhh1,
        Wh1, bh1, Wh2, bh2, out);
}
